// AE_DCRNNCell_51651276702568
// MI455X (gfx1250) — matrix operand order repaired and checked by reading, not compiled, not run on hardware
//
#include <hip/hip_runtime.h>
#include <hip/hip_bf16.h>

// ---------------------------------------------------------------------------
// AE_DCRNNCell on MI455X (gfx1250, wave32).
//  - support diffusion + per-node adaptive conv in bf16 WMMA
//    (v_wmma_f32_16x16x32_bf16), f32 accumulation
//  - per-node weight generation via f32 WMMA (v_wmma_f32_16x16x4_f32)
//  - LDS tile staging via Tensor Data Mover (tensor_load_to_lds +
//    s_wait_tensorcnt) when the builtin is available; manual staging fallback
//  Scratch requirement: ~219 MB in d_ws.
// ---------------------------------------------------------------------------

typedef __attribute__((ext_vector_type(16))) __bf16 v16bf;
typedef __attribute__((ext_vector_type(8)))  __bf16 v8bf;
typedef __attribute__((ext_vector_type(8)))  float  v8f;
typedef __attribute__((ext_vector_type(2)))  float  v2f;
typedef __attribute__((ext_vector_type(4)))  unsigned int v4u;
typedef __attribute__((ext_vector_type(8)))  int    v8i;
typedef __attribute__((ext_vector_type(4)))  int    v4i;

#define N_NODES 1024
#define BATCH   64
#define HDIM    64
#define IFEAT   66          // D_IN + H
#define NMX     7
#define KI      462         // NMX * IFEAT
#define KIP     480         // padded to 15*32
#define EDIM    16
#define COLS    4224        // BATCH * IFEAT
#define PLANE   ((size_t)N_NODES * COLS)

#if defined(__has_builtin)
#  if __has_builtin(__builtin_amdgcn_tensor_load_to_lds) && \
      __has_builtin(__builtin_amdgcn_s_wait_tensorcnt)
#    define HAVE_TDM 1
#  endif
#endif
#ifndef HAVE_TDM
#  define HAVE_TDM 0
#endif

// ---- WMMA helpers ---------------------------------------------------------

__device__ __forceinline__ v8f wmma_bf16(v16bf a, v16bf b, v8f c) {
  return __builtin_amdgcn_wmma_f32_16x16x32_bf16(false, a, false, b,
                                                 (short)0, c, false, false);
}

// A fragment (16x32 bf16) from LDS, row-major, stride in halves (mult of 8).
__device__ __forceinline__ v16bf load_frag_A(const __bf16* lds, int row0,
                                             int k0, int stride) {
  int lane = threadIdx.x & 31;
  int m  = lane & 15;
  int kb = k0 + ((lane >> 4) << 3);
  const __bf16* p = lds + (size_t)(row0 + m) * stride + kb;
  v8bf lo = *(const v8bf*)p;         // k = kb .. kb+7
  v8bf hi = *(const v8bf*)(p + 16);  // k = kb+16 .. kb+23
  return __builtin_shufflevector(lo, hi, 0,1,2,3,4,5,6,7,8,9,10,11,12,13,14,15);
}

// B fragment (32x16 bf16) from LDS stored [k][n] (stride halves).
__device__ __forceinline__ v16bf load_frag_B(const __bf16* lds, int col0,
                                             int stride) {
  int lane = threadIdx.x & 31;
  int n  = col0 + (lane & 15);
  int kb = (lane >> 4) << 3;
  v16bf f;
#pragma unroll
  for (int i = 0; i < 16; ++i) f[i] = lds[(size_t)(kb + i + ((i >> 3) << 3)) * stride + n];
  return f;
}

// ---- Tensor Data Mover: rectangular tile -> LDS with row padding ----------
// tile0e8 = row length in 8-byte units, tile1 = #rows (1 => contiguous 1D),
// stride0e8 = global row stride (8B units). LDS gets padAmt+1 dwords of pad
// after each 2^(padIntv+1) dwords (== one row), matching our LDS strides.
#if HAVE_TDM
__device__ __forceinline__ void tdm_issue(v4u g0, v8i g1) {
  v4i z4 = {0, 0, 0, 0};
#if __clang_major__ >= 23
  v8i z8 = {0, 0, 0, 0, 0, 0, 0, 0};
  __builtin_amdgcn_tensor_load_to_lds(g0, g1, z4, z4, z8, 0);
#else
  __builtin_amdgcn_tensor_load_to_lds(g0, g1, z4, z4, 0);
#endif
}

__device__ __forceinline__ void tdm_load_tile(unsigned ldsOff, const void* gptr,
                                              unsigned tile0e8, unsigned tile1,
                                              unsigned stride0e8,
                                              unsigned padIntv, unsigned padAmt) {
  unsigned long long ga = (unsigned long long)(__SIZE_TYPE__)gptr;
  v4u g0;
  g0[0] = 1u;                                                  // count=1
  g0[1] = ldsOff;                                              // lds byte addr
  g0[2] = (unsigned)ga;                                        // gaddr[31:0]
  g0[3] = (unsigned)((ga >> 32) & 0x1FFFFFFu) | (2u << 30);    // [56:32]|type=2
  v8i g1;
  g1[0] = (int)((3u << 16) | (1u << 20) | (padIntv << 22) | (padAmt << 25));
  g1[1] = (int)((tile0e8 & 0xFFFFu) << 16);                    // tensor_dim0 lo
  g1[2] = (int)((tile0e8 >> 16) | ((tile1 & 0xFFFFu) << 16));  // d0 hi | d1 lo
  g1[3] = (int)((tile1 >> 16) | ((tile0e8 & 0xFFFFu) << 16));  // d1 hi | tile0
  g1[4] = (int)(tile1 & 0xFFFFu);                              // tile_dim1
  g1[5] = (int)stride0e8;                                      // d0 stride lo
  g1[6] = 0;
  g1[7] = 0;
  tdm_issue(g0, g1);
}
#endif

// ---- small kernels --------------------------------------------------------

__global__ void k_cvt_bf16(const float* __restrict__ s, __bf16* __restrict__ d,
                           int n) {
  int i = blockIdx.x * blockDim.x + threadIdx.x;
  int st = gridDim.x * blockDim.x;
  for (; i < n; i += st) d[i] = (__bf16)s[i];
}

// nv = tanh(nv1@w1+b1) * sigmoid(nv2@w2+b2)   [1024,16]
__global__ void k_nv(const float* __restrict__ nv1, const float* __restrict__ nv2,
                     const float* __restrict__ w1, const float* __restrict__ b1,
                     const float* __restrict__ w2, const float* __restrict__ b2,
                     float* __restrict__ nv) {
  int n = blockIdx.x * blockDim.x + threadIdx.x;
  if (n >= N_NODES) return;
#pragma unroll
  for (int e = 0; e < EDIM; ++e) {
    float s1 = b1[e], s2 = b2[e];
#pragma unroll
    for (int d = 0; d < EDIM; ++d) {
      s1 += nv1[n * EDIM + d] * w1[d * EDIM + e];
      s2 += nv2[n * EDIM + d] * w2[d * EDIM + e];
    }
    nv[n * EDIM + e] = tanhf(s1) * (1.f / (1.f + __expf(-s2)));
  }
}

// adp = softmax(relu(nv1 @ nv2^T)) rows, stored bf16.  One block per row.
__global__ __launch_bounds__(256) void k_adp(const float* __restrict__ nv1,
                                             const float* __restrict__ nv2,
                                             __bf16* __restrict__ adp) {
  __shared__ float buf[N_NODES];
  __shared__ float red[256];
  int row = blockIdx.x, t = threadIdx.x;
  float a[EDIM];
#pragma unroll
  for (int d = 0; d < EDIM; ++d) a[d] = nv1[row * EDIM + d];
  float lmax = 0.f;
  for (int j = t; j < N_NODES; j += 256) {
    float dot = 0.f;
#pragma unroll
    for (int d = 0; d < EDIM; ++d) dot += a[d] * nv2[j * EDIM + d];
    float v = fmaxf(dot, 0.f);
    buf[j] = v;
    lmax = fmaxf(lmax, v);
  }
  red[t] = lmax; __syncthreads();
  for (int s = 128; s > 0; s >>= 1) {
    if (t < s) red[t] = fmaxf(red[t], red[t + s]);
    __syncthreads();
  }
  float m = red[0]; __syncthreads();
  float lsum = 0.f;
  for (int j = t; j < N_NODES; j += 256) {
    float e = __expf(buf[j] - m);
    buf[j] = e; lsum += e;
  }
  red[t] = lsum; __syncthreads();
  for (int s = 128; s > 0; s >>= 1) {
    if (t < s) red[t] += red[t + s];
    __syncthreads();
  }
  float inv = 1.f / red[0];
  for (int j = t; j < N_NODES; j += 256)
    adp[(size_t)row * N_NODES + j] = (__bf16)(buf[j] * inv);
}

// F0[n, b*66+i] = concat(x_in, h) in bf16; also x_in part of F0c.
__global__ void k_build_f0(const float* __restrict__ x, const float* __restrict__ h,
                           __bf16* __restrict__ F0, __bf16* __restrict__ F0c) {
  int i = blockIdx.x * blockDim.x + threadIdx.x;
  int st = gridDim.x * blockDim.x;
  const int TOT = N_NODES * BATCH * IFEAT;
  for (; i < TOT; i += st) {
    int n = i / (BATCH * IFEAT);
    int rem = i - n * (BATCH * IFEAT);
    int b = rem / IFEAT;
    int c = rem - b * IFEAT;
    float v = (c < 2) ? x[b * 2048 + n * 2 + c]
                      : h[(size_t)b * 65536 + n * HDIM + (c - 2)];
    __bf16 bv = (__bf16)v;
    F0[i] = bv;
    if (c < 2) F0c[i] = bv;  // x_in part of candidate features (h-part later)
  }
}

// ---- per-node weight generation: W[n, ki, o] = sum_d nv[n,d] wg[d, ki, o] --
// f32 WMMA 16x16x4, output bf16 padded [N][480*O]; pad rows written as zero.
__global__ __launch_bounds__(256) void k_wgen(const float* __restrict__ nv,
                                              const float* __restrict__ wg,
                                              __bf16* __restrict__ Wout, int O) {
  const int colsReal = KI * O;
  const int colsPad  = KIP * O;
  const int nb = blockIdx.x;                       // 0..63 -> 16 nodes
  const int wave = threadIdx.x >> 5, lane = threadIdx.x & 31;
  const int c0 = (blockIdx.y * 8 + wave) * 16;     // column tile
  const int colb = lane & 15;
  const int kgrp = (lane >> 4) << 1;               // 0 or 2
  v8f acc = {0.f, 0.f, 0.f, 0.f, 0.f, 0.f, 0.f, 0.f};
  if (c0 < colsReal) {                             // uniform per wave: EXEC all-1
    int nrow = nb * 16 + (lane & 15);
#pragma unroll
    for (int k0 = 0; k0 < EDIM; k0 += 4) {
      v2f a, b;
      a.x = nv[nrow * EDIM + k0 + kgrp];
      a.y = nv[nrow * EDIM + k0 + kgrp + 1];
      b.x = wg[(size_t)(k0 + kgrp) * colsReal + c0 + colb];
      b.y = wg[(size_t)(k0 + kgrp + 1) * colsReal + c0 + colb];
      acc = __builtin_amdgcn_wmma_f32_16x16x4_f32(false, a, false, b,
                                                  (short)0, acc, false, false);
    }
  }
  int rbase = (lane >> 4) << 3;
#pragma unroll
  for (int r = 0; r < 8; ++r) {
    int nrow = nb * 16 + rbase + r;
    Wout[(size_t)nrow * colsPad + c0 + colb] = (__bf16)acc[r];
  }
}

// ---- support GEMM: C[1024,4224] = S[1024,1024] x F[1024,4224] (bf16) -------
// Block tile 128x128, 8 waves (2x4 subtiles each), K-loop 32. Tiles staged to
// LDS by the Tensor Data Mover (double-buffered), fallback: manual staging.
// If sub!=null: C = 2*acc - sub   (Chebyshev p2 = 2*S*p1 - p0).
__global__ __launch_bounds__(256) void k_gemm_sxf(const __bf16* __restrict__ A,
                                                  const __bf16* __restrict__ B,
                                                  __bf16* __restrict__ C,
                                                  const __bf16* __restrict__ sub) {
  __shared__ __bf16 As[2][128 * 40];   // row stride 40 halves (32 + 8 pad)
  __shared__ __bf16 Bs[2][32 * 136];   // row stride 136 halves (128 + 8 pad)
  const int tid = threadIdx.x;
  const int mb = blockIdx.x, nb = blockIdx.y;
  const int wave = tid >> 5, lane = tid & 31;
  const int wm = wave & 3, wn = wave >> 2;
  v8f z = {0.f, 0.f, 0.f, 0.f, 0.f, 0.f, 0.f, 0.f};
  v8f acc[2][4] = {{z, z, z, z}, {z, z, z, z}};

#if HAVE_TDM
  const __bf16* Abase = A + (size_t)mb * 128 * 1024;
  const __bf16* Bbase = B + (size_t)nb * 128;
  const unsigned aOff[2] = {(unsigned)(__SIZE_TYPE__)(void*)&As[0][0],
                            (unsigned)(__SIZE_TYPE__)(void*)&As[1][0]};
  const unsigned bOff[2] = {(unsigned)(__SIZE_TYPE__)(void*)&Bs[0][0],
                            (unsigned)(__SIZE_TYPE__)(void*)&Bs[1][0]};
  if (wave == 0) {  // A: 128 rows x 64B, stride 2048B. B: 32 rows x 256B, stride 8448B.
    tdm_load_tile(aOff[0], Abase, 8, 128, 256, 3, 3);
    tdm_load_tile(bOff[0], Bbase, 32, 32, 1056, 5, 3);
  }
  for (int it = 0; it < 32; ++it) {
    const int cur = it & 1;
    __syncthreads();                    // prior compute on buf cur^1 done
    if (wave == 0) {
      if (it < 31) {
        tdm_load_tile(aOff[cur ^ 1], Abase + (it + 1) * 32, 8, 128, 256, 3, 3);
        tdm_load_tile(bOff[cur ^ 1], Bbase + (size_t)(it + 1) * 32 * COLS,
                      32, 32, 1056, 5, 3);
        __builtin_amdgcn_s_wait_tensorcnt(2);   // pair(it) complete (in-order)
      } else {
        __builtin_amdgcn_s_wait_tensorcnt(0);
      }
    }
    __syncthreads();
    v16bf af0 = load_frag_A(&As[cur][0], wm * 32,      0, 40);
    v16bf af1 = load_frag_A(&As[cur][0], wm * 32 + 16, 0, 40);
#pragma unroll
    for (int j = 0; j < 4; ++j) {
      v16bf bf = load_frag_B(&Bs[cur][0], wn * 64 + j * 16, 136);
      acc[0][j] = wmma_bf16(af0, bf, acc[0][j]);
      acc[1][j] = wmma_bf16(af1, bf, acc[1][j]);
    }
  }
#else
  const int ra = tid & 127, sega = (tid >> 7) << 4;
  const int rb = tid >> 3, cb = (tid & 7) << 4;
  const __bf16* Ap = A + (size_t)(mb * 128 + ra) * 1024 + sega;
  const __bf16* Bp = B + (size_t)rb * COLS + nb * 128 + cb;
  v8bf a0 = *(const v8bf*)Ap,        a1 = *(const v8bf*)(Ap + 8);
  v8bf b0 = *(const v8bf*)Bp,        b1 = *(const v8bf*)(Bp + 8);
  *(v8bf*)&As[0][ra * 40 + sega]     = a0;
  *(v8bf*)&As[0][ra * 40 + sega + 8] = a1;
  *(v8bf*)&Bs[0][rb * 136 + cb]      = b0;
  *(v8bf*)&Bs[0][rb * 136 + cb + 8]  = b1;
  for (int it = 0; it < 32; ++it) {
    __syncthreads();
    if (it < 31) {
      const __bf16* Ag = Ap + (it + 1) * 32;
      const __bf16* Bg = Bp + (size_t)(it + 1) * 32 * COLS;
      a0 = *(const v8bf*)Ag; a1 = *(const v8bf*)(Ag + 8);
      b0 = *(const v8bf*)Bg; b1 = *(const v8bf*)(Bg + 8);
    }
    const int cur = it & 1;
    v16bf af0 = load_frag_A(&As[cur][0], wm * 32,      0, 40);
    v16bf af1 = load_frag_A(&As[cur][0], wm * 32 + 16, 0, 40);
#pragma unroll
    for (int j = 0; j < 4; ++j) {
      v16bf bf = load_frag_B(&Bs[cur][0], wn * 64 + j * 16, 136);
      acc[0][j] = wmma_bf16(af0, bf, acc[0][j]);
      acc[1][j] = wmma_bf16(af1, bf, acc[1][j]);
    }
    if (it < 31) {
      const int nxt = cur ^ 1;
      *(v8bf*)&As[nxt][ra * 40 + sega]     = a0;
      *(v8bf*)&As[nxt][ra * 40 + sega + 8] = a1;
      *(v8bf*)&Bs[nxt][rb * 136 + cb]      = b0;
      *(v8bf*)&Bs[nxt][rb * 136 + cb + 8]  = b1;
    }
  }
#endif

  const int colb = lane & 15, rowb = (lane >> 4) << 3;
#pragma unroll
  for (int i = 0; i < 2; ++i)
#pragma unroll
    for (int j = 0; j < 4; ++j) {
      int c = nb * 128 + wn * 64 + j * 16 + colb;
#pragma unroll
      for (int r = 0; r < 8; ++r) {
        int m = mb * 128 + wm * 32 + i * 16 + rowb + r;
        size_t idx = (size_t)m * COLS + c;
        float v = acc[i][j][r];
        if (sub) v = 2.f * v - (float)sub[idx];
        C[idx] = (__bf16)v;
      }
    }
}

// ---- per-node adaptive GEMM + fused GRU epilogue ---------------------------
// One block per node. X_n [64 x 480] staged in LDS from the 7 planes; W_n
// streamed in 32-row slices (TDM double-buffered). MODE 0: ru (sigmoid, split
// r/u, emit r*h into F0c and u into u_buf). MODE 1: candidate (tanh, GRU out).
template <int O, int MODE>
__global__ __launch_bounds__(256) void k_pernode(const __bf16* __restrict__ planes,
                                                 const __bf16* __restrict__ Wall,
                                                 const float* __restrict__ nv,
                                                 const float* __restrict__ bg,
                                                 const float* __restrict__ state,
                                                 float* __restrict__ u_buf,
                                                 __bf16* __restrict__ F0c,
                                                 float* __restrict__ outp) {
  __shared__ __bf16 Xs[BATCH * KIP];   // 61.4 KB
  __shared__ __bf16 Ws[2][32 * 136];   // 17.4 KB (O+8 row stride <= 136)
  const int n = blockIdx.x;
  const int tid = threadIdx.x;
  const int wave = tid >> 5, lane = tid & 31;
  constexpr int NT = O >> 4;           // 8 (ru) or 4 (can) col tiles
  constexpr int JT = NT >> 1;          // tiles per wave group
  constexpr int WSTR = O + 8;
  const int mtile = wave & 3;
  const int grp = wave >> 2;
  const __bf16* Wn = Wall + (size_t)n * KIP * O;

  for (int idx = tid; idx < BATCH * KIP; idx += 256) {
    int b = idx / KIP, k = idx - b * KIP;
    __bf16 v = (__bf16)0.f;
    if (k < KI) {
      int ks = k / IFEAT, i = k - ks * IFEAT;
      v = planes[(size_t)ks * PLANE + (size_t)n * COLS + b * IFEAT + i];
    }
    Xs[idx] = v;
  }

  v8f z = {0.f, 0.f, 0.f, 0.f, 0.f, 0.f, 0.f, 0.f};
  v8f acc[JT];
#pragma unroll
  for (int j = 0; j < JT; ++j) acc[j] = z;

#if HAVE_TDM
  constexpr unsigned PI = (O == 128) ? 5u : 4u;  // LDS pad every row
  const unsigned wOff[2] = {(unsigned)(__SIZE_TYPE__)(void*)&Ws[0][0],
                            (unsigned)(__SIZE_TYPE__)(void*)&Ws[1][0]};
  if (wave == 0)  // slice 0: contiguous 32*O halves (8*O qwords)
    tdm_load_tile(wOff[0], Wn, 8 * O, 1, 8 * O, PI, 3);
#endif
  __syncthreads();   // Xs ready

  int itc = 0;
  for (int kk = 0; kk < KIP; kk += 32, ++itc) {
    const int cur = itc & 1;
#if HAVE_TDM
    if (wave == 0) {
      if (kk + 32 < KIP) {
        tdm_load_tile(wOff[cur ^ 1], Wn + (size_t)(kk + 32) * O, 8 * O, 1,
                      8 * O, PI, 3);
        __builtin_amdgcn_s_wait_tensorcnt(1);   // slice(kk) complete
      } else {
        __builtin_amdgcn_s_wait_tensorcnt(0);
      }
    }
    __syncthreads();
#else
    if (kk + 32 < KIP)
      __builtin_prefetch((const void*)(Wn + (size_t)(kk + 32) * O + tid * 16), 0, 1);
    constexpr int ELEMS8 = (32 * O) >> 3;
    for (int e = tid; e < ELEMS8; e += 256) {
      int off = e << 3;
      int r = off / O, c = off - r * O;
      *(v8bf*)&Ws[cur][r * WSTR + c] = *(const v8bf*)(Wn + (size_t)(kk + r) * O + c);
    }
    __syncthreads();
#endif
    v16bf af = load_frag_A(Xs, mtile * 16, kk, KIP);
#pragma unroll
    for (int j = 0; j < JT; ++j) {
      v16bf bf = load_frag_B(&Ws[cur][0], (grp * JT + j) * 16, WSTR);
      acc[j] = wmma_bf16(af, bf, acc[j]);
    }
    __syncthreads();   // before buf cur is overwritten next time
  }

  const int colb = lane & 15, rowb = (lane >> 4) << 3;
#pragma unroll
  for (int j = 0; j < JT; ++j) {
    int col = (grp * JT + j) * 16 + colb;           // output channel
    float bias = 0.f;
#pragma unroll
    for (int d = 0; d < EDIM; ++d) bias += nv[n * EDIM + d] * bg[d * O + col];
#pragma unroll
    for (int r = 0; r < 8; ++r) {
      int b = mtile * 16 + rowb + r;
      float val = acc[j][r] + bias;
      if (MODE == 0) {
        float s = 1.f / (1.f + __expf(-val));
        if (col < HDIM) {  // r gate -> r*h becomes candidate-phase feature
          float rh = s * state[(size_t)b * 65536 + n * HDIM + col];
          F0c[(size_t)n * COLS + b * IFEAT + 2 + col] = (__bf16)rh;
        } else {           // u gate
          u_buf[(size_t)n * 4096 + b * HDIM + (col - HDIM)] = s;
        }
      } else {
        float c = tanhf(val);
        float u = u_buf[(size_t)n * 4096 + b * HDIM + col];
        float h = state[(size_t)b * 65536 + n * HDIM + col];
        outp[(size_t)b * 65536 + n * HDIM + col] = u * h + (1.f - u) * c;
      }
    }
  }
}

// ---------------------------------------------------------------------------

extern "C" void kernel_launch(void* const* d_in, const int* in_sizes, int n_in,
                              void* d_out, int out_size, void* d_ws, size_t ws_size,
                              hipStream_t stream) {
  (void)in_sizes; (void)n_in; (void)out_size; (void)ws_size;
  const float* inputs = (const float*)d_in[0];
  const float* state  = (const float*)d_in[1];
  const float* nv1    = (const float*)d_in[2];
  const float* nv2    = (const float*)d_in[3];
  const float* w1     = (const float*)d_in[4];
  const float* b1     = (const float*)d_in[5];
  const float* w2     = (const float*)d_in[6];
  const float* b2     = (const float*)d_in[7];
  const float* ru_wg  = (const float*)d_in[8];
  const float* ru_bg  = (const float*)d_in[9];
  const float* can_wg = (const float*)d_in[10];
  const float* can_bg = (const float*)d_in[11];
  const float* sup0   = (const float*)d_in[12];
  const float* sup1   = (const float*)d_in[13];
  float* outp = (float*)d_out;

  // workspace carve (~219 MB)
  char* base = (char*)d_ws;
  size_t off = 0;
  auto take = [&](size_t bytes) -> void* {
    void* p = base + off;
    off = (off + bytes + 255) & ~(size_t)255;
    return p;
  };
  __bf16* S0b  = (__bf16*)take((size_t)N_NODES * N_NODES * 2);
  __bf16* S1b  = (__bf16*)take((size_t)N_NODES * N_NODES * 2);
  __bf16* ADPb = (__bf16*)take((size_t)N_NODES * N_NODES * 2);
  __bf16* P    = (__bf16*)take((size_t)NMX * PLANE * 2);   // 7 contiguous planes
  __bf16* F0c  = (__bf16*)take(PLANE * 2);
  float*  nvb  = (float*)take((size_t)N_NODES * EDIM * 4);
  float*  ubuf = (float*)take((size_t)N_NODES * BATCH * HDIM * 4);
  __bf16* Wall = (__bf16*)take((size_t)N_NODES * KIP * 128 * 2);  // reused for can

  const __bf16* Sb[3] = {S0b, S1b, ADPb};

  // precompute stage
  k_cvt_bf16<<<2048, 256, 0, stream>>>(sup0, S0b, N_NODES * N_NODES);
  k_cvt_bf16<<<2048, 256, 0, stream>>>(sup1, S1b, N_NODES * N_NODES);
  k_adp<<<N_NODES, 256, 0, stream>>>(nv1, nv2, ADPb);
  k_nv<<<4, 256, 0, stream>>>(nv1, nv2, w1, b1, w2, b2, nvb);
  k_build_f0<<<2048, 256, 0, stream>>>(inputs, state, P, F0c);

  // ---- reset/update gates phase ----
  k_wgen<<<dim3(64, 480), 256, 0, stream>>>(nvb, ru_wg, Wall, 128);
  for (int s = 0; s < 3; ++s)   // p1 = S x0
    k_gemm_sxf<<<dim3(8, 33), 256, 0, stream>>>(Sb[s], P, P + (size_t)(1 + 2 * s) * PLANE, nullptr);
  for (int s = 0; s < 3; ++s)   // p2 = 2 S p1 - x0
    k_gemm_sxf<<<dim3(8, 33), 256, 0, stream>>>(Sb[s], P + (size_t)(1 + 2 * s) * PLANE,
                                                P + (size_t)(2 + 2 * s) * PLANE, P);
  k_pernode<128, 0><<<N_NODES, 256, 0, stream>>>(P, Wall, nvb, ru_bg, state,
                                                 ubuf, F0c, nullptr);

  // candidate features: plane0 <- concat(x_in, r*h)
  hipMemcpyAsync(P, F0c, PLANE * 2, hipMemcpyDeviceToDevice, stream);

  // ---- candidate phase ----
  k_wgen<<<dim3(64, 240), 256, 0, stream>>>(nvb, can_wg, Wall, 64);
  for (int s = 0; s < 3; ++s)
    k_gemm_sxf<<<dim3(8, 33), 256, 0, stream>>>(Sb[s], P, P + (size_t)(1 + 2 * s) * PLANE, nullptr);
  for (int s = 0; s < 3; ++s)
    k_gemm_sxf<<<dim3(8, 33), 256, 0, stream>>>(Sb[s], P + (size_t)(1 + 2 * s) * PLANE,
                                                P + (size_t)(2 + 2 * s) * PLANE, P);
  k_pernode<64, 1><<<N_NODES, 256, 0, stream>>>(P, Wall, nvb, can_bg, state,
                                                ubuf, F0c, outp);
}
